// C3PartialConv_7301444403578
// MI455X (gfx1250) — matrix operand order repaired and checked by reading, not compiled, not run on hardware
//
#include <hip/hip_runtime.h>

typedef __attribute__((ext_vector_type(16))) __bf16        v16bf;
typedef __attribute__((ext_vector_type(8)))  float         v8f;
typedef __attribute__((ext_vector_type(8)))  unsigned int  v8u;

#define ROWS   6
#define LDS_W  132
#define LDS_H  (ROWS + 4)
#define KCH    5          // 5 chunks of K=32 -> K=160 (150 real + 10 zero-A pad)
#define NPLANE (LDS_H * (LDS_W / 2))   // dwords per channel plane (660)

// 6-bit connectivity mask per output channel (C3_CONNECTIONS baked in)
__constant__ unsigned char kC3Mask[16] = {
  0x07, 0x0E, 0x1C, 0x38, 0x31, 0x23,
  0x0F, 0x1E, 0x3C, 0x39, 0x33, 0x27,
  0x1B, 0x36, 0x2D, 0x3F
};

static __device__ __forceinline__ unsigned short f2bf_rne(float f) {
  unsigned int u = __float_as_uint(f);
  u += 0x7FFFu + ((u >> 16) & 1u);          // round-to-nearest-even
  return (unsigned short)(u >> 16);
}
// LDS element offset of im2col index k (k = ic*25 + ky*5 + kx).
// k in [150,175) lands in the junk channel 6 (A is zero there).
static constexpr int off_nat(int k) {
  return ((k / 25) * LDS_H + (k % 25) / 5) * LDS_W + (k % 5);
}
// The half-displacement off(k+16)-off(k) takes only 4 distinct values:
static constexpr int D_A = 3 * LDS_W + 1;            // (ky+3, kx+1)
static constexpr int D_B = 4 * LDS_W - 4;            // (ky+4, kx-4)
static constexpr int D_C = (LDS_H - 1) * LDS_W - 4;  // ic+1 row-wrap, kx-4
static constexpr int D_D = (LDS_H - 2) * LDS_W + 1;  // ic+1 row-wrap, kx+1

extern "C" __global__ __launch_bounds__(256)
void c3_wmma_kernel(const float* __restrict__ x,
                    const float* __restrict__ W,
                    const float* __restrict__ bias,
                    float* __restrict__ out)
{
  // 7 channels x 10 rows x 132 bf16; channel 6 is never written (zero-A junk)
  __shared__ __align__(16) unsigned short s_x[7 * LDS_H * LDS_W];   // 18480 B
  // prebuilt A fragments: [chunk][lane][16 bf16]  (5120 B)
  __shared__ __align__(16) unsigned short s_af[KCH * 32 * 16];

  const int tx  = blockIdx.x;           // 0..1 : 128-wide column tiles
  const int oy0 = blockIdx.y * ROWS;    // 0..246
  const int b   = blockIdx.z;           // batch
  const int tid = threadIdx.x;
  const int x0  = tx * 128;

  unsigned int* s_x32 = (unsigned int*)s_x;

  // ---- stage input patch as bf16 pairs: float2 load -> packed dword ----
  {
    const int NPAIR = 6 * NPLANE;                  // 3960 dwords of real data
    for (int pd = tid; pd < NPAIR; pd += 256) {
      const int jp  = pd % (LDS_W / 2);
      const int t   = pd / (LDS_W / 2);
      const int row = t % LDS_H;
      const int ic  = t / LDS_H;
      const int gx  = x0 + jp * 2;                 // even -> float2 aligned
      float2 v = make_float2(0.0f, 0.0f);
      if (gx < 256)                                // pair never straddles 256
        v = *(const float2*)(x + (((b * 6 + ic) * 256 + (oy0 + row)) * 256 + gx));
      const unsigned int u0 = __float_as_uint(v.x) + 0x8000u;  // cheap round
      const unsigned int u1 = __float_as_uint(v.y) + 0x8000u;
      s_x32[pd] = (u1 & 0xFFFF0000u) | (u0 >> 16);
    }
  }

  // ---- build masked-weight A fragments ONCE per block (all threads) ----
  for (int e = tid; e < KCH * 32 * 16; e += 256) {
    const int j    = e & 15;
    const int ln   = (e >> 4) & 31;
    const int c    = e >> 9;
    const int ehh  = ln >> 4;
    const int emn  = ln & 15;
    // k order used here (contiguous, same as B): half 0 holds K = c*32 + 0..15; half 1: +16
    const int k = c * 32 + j + 16 * ehh;
    float v = 0.0f;
    if (k < 150) {
      const int ic = k / 25;
      if ((kC3Mask[emn] >> ic) & 1u) v = W[emn * 150 + k];
    }
    s_af[e] = f2bf_rne(v);
  }

  __syncthreads();

  const int lane = tid & 31;
  const int wave = tid >> 5;
  const int hh   = lane >> 4;           // lane half (ISA VGPR layout selector)
  const int mn   = lane & 15;           // M for A / N for B,C,D

  // fetch this lane's A fragments: 2x ds_load_b128 per chunk
  v16bf afrag[KCH];
  {
    const v8u* ap = (const v8u*)s_af;
    #pragma unroll
    for (int c = 0; c < KCH; ++c)
      afrag[c] = __builtin_bit_cast(v16bf, ap[c * 32 + lane]);
  }

  // pre-scaled bias for this lane's 8 C-tile rows (M = e + 8*hh)
  float bz[8];
  #pragma unroll
  for (int e = 0; e < 8; ++e) bz[e] = (2.0f / 3.0f) * bias[e + 8 * hh];

  const int colbase = wave * 16 + mn;   // LDS column of this lane's B column
  const int ocol    = x0 + colbase;     // global output column

  // ---- the ONLY per-lane gather state: 4 base registers ----
  const int baseA = colbase + hh * D_A;
  const int baseB = colbase + hh * D_B;
  const int baseC = colbase + hh * D_C;
  const int baseD = colbase + hh * D_D;

  // gather one bf16: k0 and r are compile-time -> bare ds_load base+imm
  auto gat = [&](int k0, int r) -> unsigned int {
    const int d = off_nat(k0 + 16) - off_nat(k0);
    const int base = (d == D_A) ? baseA
                   : (d == D_B) ? baseB
                   : (d == D_C) ? baseC
                   :              baseD;
    return (unsigned int)s_x[base + off_nat(k0) + r * LDS_W];
  };

  const int obase = ((b * 16 + 8 * hh) * 252 + oy0) * 252 + ocol;

  // epilogue for one row: bias + 1.7159*tanh(2y/3), NT stores
  auto epilogue = [&](int r, const v8f& acc) {
    if (ocol < 252) {
      float* orow = out + (obase + r * 252);
      #pragma unroll
      for (int e = 0; e < 8; ++e) {
        const float z  = __builtin_fmaf(acc[e], 2.0f / 3.0f, bz[e]);
        const float t  = __builtin_amdgcn_exp2f(-2.885390082f * fabsf(z));
        const float rr = __builtin_amdgcn_rcpf(1.0f + t);
        const float s  = __builtin_fmaf(-1.7159f, t, 1.7159f);  // 1.7159*(1-t)
        const float res = copysignf(s * rr, z);
        __builtin_nontemporal_store(res, orow + e * (252 * 252));
      }
    }
  };

  // ---- row pairs: two independent accumulators per chunk so dependent
  //      WMMAs are 2 apart (absorbs WMMA data-hazard bubbles, lets the
  //      bfrag packing VALU co-execute under the matrix pipe) ----
  #pragma unroll
  for (int rp = 0; rp < ROWS / 2; ++rp) {
    const int r0 = 2 * rp, r1 = 2 * rp + 1;
    v8f acc0 = {}, acc1 = {};
    #pragma unroll
    for (int c = 0; c < KCH; ++c) {
      v8u bu0, bu1;
      #pragma unroll
      for (int t = 0; t < 8; ++t) {
        // k order used here (contiguous, same as A): lanes 0-15 hold K = c*32 + 0..15, lanes 16-31: +16
        bu0[t] = gat(c * 32 + 2 * t, r0) | (gat(c * 32 + 2 * t + 1, r0) << 16);
        bu1[t] = gat(c * 32 + 2 * t, r1) | (gat(c * 32 + 2 * t + 1, r1) << 16);
      }
      acc0 = __builtin_amdgcn_wmma_f32_16x16x32_bf16(
          false, afrag[c], false, __builtin_bit_cast(v16bf, bu0),
          (short)0, acc0, false, false);
      acc1 = __builtin_amdgcn_wmma_f32_16x16x32_bf16(
          false, afrag[c], false, __builtin_bit_cast(v16bf, bu1),
          (short)0, acc1, false, false);
    }
    epilogue(r0, acc0);
    epilogue(r1, acc1);
  }
}

extern "C" void kernel_launch(void* const* d_in, const int* in_sizes, int n_in,
                              void* d_out, int out_size, void* d_ws, size_t ws_size,
                              hipStream_t stream) {
  const float* x  = (const float*)d_in[0];
  const float* W  = (const float*)d_in[1];
  const float* bb = (const float*)d_in[2];
  float* out = (float*)d_out;

  dim3 grid(2, 252 / ROWS, 64);   // col-tile, row-block, batch
  dim3 block(256);                // 8 wave32 per block
  hipLaunchKernelGGL(c3_wmma_kernel, grid, block, 0, stream, x, W, bb, out);
}
